// ECGGATModel_24154896073072
// MI455X (gfx1250) — matrix operand order repaired and checked by reading, not compiled, not run on hardware
//
#include <hip/hip_runtime.h>
#include <hip/hip_bf16.h>

// ---------------------------------------------------------------------------
// Fused GAT (2 layers + pool + MLP head) for MI455X / gfx1250, bf16 WMMA.
//   B=32768 graphs, N=12 nodes (padded to 16), F_IN=128, H=4 heads, C=64.
// One workgroup (128 threads = 4 wave32) per graph; one wave per head.
// Weights are pre-packed to bf16 WMMA B-fragment order in d_ws by a prep
// kernel so the hot loop does contiguous 128-bit loads.
// ---------------------------------------------------------------------------

typedef __attribute__((ext_vector_type(16))) __bf16 v16bf;
typedef __attribute__((ext_vector_type(8)))  float  v8f;
typedef __bf16 bf16_t;

union FragAB { v16bf v; bf16_t e[16]; };

__device__ __forceinline__ float warp_sum32(float v) {
#pragma unroll
  for (int o = 16; o > 0; o >>= 1) v += __shfl_xor(v, o, 32);
  return v;
}
__device__ __forceinline__ float warp_max32(float v) {
#pragma unroll
  for (int o = 16; o > 0; o >>= 1) v = fmaxf(v, __shfl_xor(v, o, 32));
  return v;
}
__device__ __forceinline__ float elu_f(float x) {
  return x > 0.f ? x : (__expf(x) - 1.f);
}
__device__ __forceinline__ float lrelu_f(float x) {
  return x > 0.f ? x : 0.2f * x;
}

// ---------------------------------------------------------------------------
// Pack W1 [128,4,64] and W2 [256,1,64] (f32) into bf16 WMMA B-fragments.
// B-frag order here (32x16; contiguous, not AMD's table order): lane l -> column n = l%16, K run of 16
// contiguous values starting at 16*(l/16); element order within the lane's
// 16 bf16 = K ascending (matches VGPR dword pairs K=2v,2v+1).
// Fragment index W1: ((h*4 + ks)*4 + t);   W2: (ks*4 + t).
// ---------------------------------------------------------------------------
__global__ void pack_weights_kernel(const float* __restrict__ W1,
                                    const float* __restrict__ W2,
                                    bf16_t* __restrict__ w1p,
                                    bf16_t* __restrict__ w2p) {
  const int stride = gridDim.x * blockDim.x;
  int idx = blockIdx.x * blockDim.x + threadIdx.x;
  // W1: 4 heads * 4 ksteps * 4 ntiles = 64 frags * 512 bf16
  for (int i = idx; i < 64 * 512; i += stride) {
    const int e    = i & 15;
    const int lane = (i >> 4) & 31;
    const int frag = i >> 9;
    const int t  = frag & 3;
    const int ks = (frag >> 2) & 3;
    const int h  = frag >> 4;
    const int n  = t * 16 + (lane & 15);
    const int k  = ks * 32 + 16 * (lane >> 4) + e;
    w1p[i] = (bf16_t)W1[k * 256 + h * 64 + n];   // W1[k][h][n]
  }
  // W2: 8 ksteps * 4 ntiles = 32 frags * 512 bf16
  for (int i = idx; i < 32 * 512; i += stride) {
    const int e    = i & 15;
    const int lane = (i >> 4) & 31;
    const int frag = i >> 9;
    const int t  = frag & 3;
    const int ks = frag >> 2;
    const int n  = t * 16 + (lane & 15);
    const int k  = ks * 32 + 16 * (lane >> 4) + e;
    w2p[i] = (bf16_t)W2[k * 64 + n];             // W2[k][0][n]
  }
}

// ---------------------------------------------------------------------------
// Fused per-graph kernel.
// ---------------------------------------------------------------------------
__global__ __launch_bounds__(128)
void gat_fused_kernel(const float* __restrict__ x,        // [B,12,128]
                      const bf16_t* __restrict__ w1p,     // packed bf16
                      const bf16_t* __restrict__ w2p,     // packed bf16
                      const float* __restrict__ a_src1,   // [4,64]
                      const float* __restrict__ a_dst1,   // [4,64]
                      const float* __restrict__ b1,       // [256]
                      const float* __restrict__ a_src2,   // [64]
                      const float* __restrict__ a_dst2,   // [64]
                      const float* __restrict__ b2,       // [64]
                      const float* __restrict__ Wc1,      // [64,32]
                      const float* __restrict__ bc1,      // [32]
                      const float* __restrict__ Wc2,      // [32,1]
                      const float* __restrict__ bc2,      // [1]
                      float* __restrict__ out)            // [B,1]
{
  __shared__ bf16_t xbf[16][128];      // layer-1 A operand (rows 12..15 zero)
  __shared__ float  hbuf[4][16][64];   // per-head h1 (pre-attention)
  __shared__ bf16_t x2bf[16][256];     // layer-2 A operand (ELU'd layer-1 out)
  __shared__ float  h2buf[16][64];     // layer-2 h2 (pre-attention)
  __shared__ float  esrc[4][16];
  __shared__ float  edst[4][16];
  __shared__ float  alpha[4][12][16];
  __shared__ float  x3[12][64];        // layer-2 node output
  __shared__ float  gvec[64];          // pooled graph embedding

  const int b    = blockIdx.x;
  const int tid  = threadIdx.x;
  const int wave = tid >> 5;           // 0..3 == head id for layer 1
  const int lane = tid & 31;
  const int lm    = lane & 15;
  const int khalf = lane >> 4;

  // ---- stage node features into LDS as bf16, zero-pad rows 12..15 ----
  const float* xb = x + (size_t)b * (12 * 128);
  for (int i = tid; i < 16 * 128; i += 128) {
    const int r = i >> 7, c = i & 127;
    xbf[r][c] = (bf16_t)((r < 12) ? xb[r * 128 + c] : 0.f);
  }
  for (int i = tid; i < 4 * 256; i += 128) {      // pad rows of x2bf
    x2bf[12 + (i >> 8)][i & 255] = (bf16_t)0.f;
  }
  __syncthreads();

  // =========================== Layer 1 ===================================
  // h1[:, head, :] = x (16x128) @ W1[:, head, :] (128x64); 4 k-steps, 4 n-tiles
  {
    const int h = wave;
    v8f acc[4];
#pragma unroll
    for (int t = 0; t < 4; ++t) acc[t] = (v8f){0.f,0.f,0.f,0.f,0.f,0.f,0.f,0.f};

#pragma unroll
    for (int ks = 0; ks < 4; ++ks) {
      FragAB A;
      const bf16_t* rowp = &xbf[lm][ks * 32 + 16 * khalf];
#pragma unroll
      for (int e = 0; e < 16; ++e) A.e[e] = rowp[e];
#pragma unroll
      for (int t = 0; t < 4; ++t) {
        FragAB Bf;
        const bf16_t* bp = w1p + ((((h * 4 + ks) * 4 + t) * 32 + lane) * 16);
#pragma unroll
        for (int e = 0; e < 16; ++e) Bf.e[e] = bp[e];
        acc[t] = __builtin_amdgcn_wmma_f32_16x16x32_bf16(
            false, A.v, false, Bf.v, (short)0, acc[t], false, false);
      }
    }
    // spill h1 tile to LDS: element (M = r + 8*khalf, n = 16t + lm)
#pragma unroll
    for (int t = 0; t < 4; ++t)
#pragma unroll
      for (int r = 0; r < 8; ++r)
        hbuf[h][r + 8 * khalf][t * 16 + lm] = acc[t][r];
  }
  __syncthreads();

  // ---- layer-1 attention per head (wave == head) ----
  {
    const int h = wave;
    const float as0 = a_src1[h * 64 + lane], as1 = a_src1[h * 64 + 32 + lane];
    const float ad0 = a_dst1[h * 64 + lane], ad1 = a_dst1[h * 64 + 32 + lane];
    for (int j = 0; j < 12; ++j) {
      const float h0 = hbuf[h][j][lane], h1 = hbuf[h][j][32 + lane];
      const float es = warp_sum32(h0 * as0 + h1 * as1);
      const float ed = warp_sum32(h0 * ad0 + h1 * ad1);
      if (lane == 0) { esrc[h][j] = es; edst[h][j] = ed; }
    }
    for (int i = 0; i < 12; ++i) {
      float lg = -1e30f;
      if (lane < 12) lg = lrelu_f(edst[h][i] + esrc[h][lane]);
      const float m = warp_max32(lg);
      const float p = (lane < 12) ? __expf(lg - m) : 0.f;
      const float s = warp_sum32(p);
      if (lane < 12) alpha[h][i][lane] = p / s;
    }
    // aggregate + bias + ELU -> bf16 layer-2 operand
    const float bb0 = b1[h * 64 + lane], bb1 = b1[h * 64 + 32 + lane];
    for (int i = 0; i < 12; ++i) {
      float o0 = 0.f, o1 = 0.f;
      for (int j = 0; j < 12; ++j) {
        const float al = alpha[h][i][j];
        o0 += al * hbuf[h][j][lane];
        o1 += al * hbuf[h][j][32 + lane];
      }
      x2bf[i][h * 64 + lane]      = (bf16_t)elu_f(o0 + bb0);
      x2bf[i][h * 64 + 32 + lane] = (bf16_t)elu_f(o1 + bb1);
    }
  }
  __syncthreads();

  // =========================== Layer 2 ===================================
  // h2 = x2 (16x256) @ W2 (256x64); each wave owns one 16-col tile, 8 k-steps
  {
    const int t = wave;
    v8f acc = (v8f){0.f,0.f,0.f,0.f,0.f,0.f,0.f,0.f};
#pragma unroll
    for (int ks = 0; ks < 8; ++ks) {
      FragAB A;
      const bf16_t* rowp = &x2bf[lm][ks * 32 + 16 * khalf];
#pragma unroll
      for (int e = 0; e < 16; ++e) A.e[e] = rowp[e];
      FragAB Bf;
      const bf16_t* bp = w2p + (((ks * 4 + t) * 32 + lane) * 16);
#pragma unroll
      for (int e = 0; e < 16; ++e) Bf.e[e] = bp[e];
      acc = __builtin_amdgcn_wmma_f32_16x16x32_bf16(
          false, A.v, false, Bf.v, (short)0, acc, false, false);
    }
#pragma unroll
    for (int r = 0; r < 8; ++r)
      h2buf[r + 8 * khalf][t * 16 + lm] = acc[r];
  }
  __syncthreads();

  // ---- layer-2 attention, mean pool, classifier (wave 0 only) ----
  if (wave == 0) {
    const float as0 = a_src2[lane], as1 = a_src2[32 + lane];
    const float ad0 = a_dst2[lane], ad1 = a_dst2[32 + lane];
    for (int j = 0; j < 12; ++j) {
      const float h0 = h2buf[j][lane], h1 = h2buf[j][32 + lane];
      const float es = warp_sum32(h0 * as0 + h1 * as1);
      const float ed = warp_sum32(h0 * ad0 + h1 * ad1);
      if (lane == 0) { esrc[0][j] = es; edst[0][j] = ed; }
    }
    for (int i = 0; i < 12; ++i) {
      float lg = -1e30f;
      if (lane < 12) lg = lrelu_f(edst[0][i] + esrc[0][lane]);
      const float m = warp_max32(lg);
      const float p = (lane < 12) ? __expf(lg - m) : 0.f;
      const float s = warp_sum32(p);
      if (lane < 12) alpha[0][i][lane] = p / s;
    }
    const float bb0 = b2[lane], bb1 = b2[32 + lane];
    for (int i = 0; i < 12; ++i) {
      float o0 = 0.f, o1 = 0.f;
      for (int j = 0; j < 12; ++j) {
        const float al = alpha[0][i][j];
        o0 += al * h2buf[j][lane];
        o1 += al * h2buf[j][32 + lane];
      }
      x3[i][lane]      = elu_f(o0 + bb0);
      x3[i][32 + lane] = elu_f(o1 + bb1);
    }
    // global mean pool over 12 nodes
    float g0 = 0.f, g1 = 0.f;
    for (int i = 0; i < 12; ++i) { g0 += x3[i][lane]; g1 += x3[i][32 + lane]; }
    gvec[lane]      = g0 * (1.f / 12.f);
    gvec[32 + lane] = g1 * (1.f / 12.f);
    // classifier: lane m computes hdn[m] = relu(g @ Wc1[:,m] + bc1[m])
    float acc = bc1[lane];
    for (int k = 0; k < 64; ++k) acc += gvec[k] * Wc1[k * 32 + lane];
    acc = fmaxf(acc, 0.f);
    const float logit = warp_sum32(acc * Wc2[lane]) + bc2[0];
    if (lane == 0) out[b] = logit;
  }
}

// ---------------------------------------------------------------------------
extern "C" void kernel_launch(void* const* d_in, const int* in_sizes, int n_in,
                              void* d_out, int out_size, void* d_ws, size_t ws_size,
                              hipStream_t stream) {
  const float* x       = (const float*)d_in[0];
  const float* W1      = (const float*)d_in[1];
  const float* a_src1  = (const float*)d_in[2];
  const float* a_dst1  = (const float*)d_in[3];
  const float* b1      = (const float*)d_in[4];
  const float* W2      = (const float*)d_in[5];
  const float* a_src2  = (const float*)d_in[6];
  const float* a_dst2  = (const float*)d_in[7];
  const float* b2      = (const float*)d_in[8];
  const float* Wc1     = (const float*)d_in[9];
  const float* bc1     = (const float*)d_in[10];
  const float* Wc2     = (const float*)d_in[11];
  const float* bc2     = (const float*)d_in[12];
  float* out           = (float*)d_out;

  bf16_t* w1p = (bf16_t*)d_ws;                    // 64 frags * 1KB = 64 KB
  bf16_t* w2p = (bf16_t*)((char*)d_ws + 65536);   // 32 frags * 1KB = 32 KB

  pack_weights_kernel<<<64, 256, 0, stream>>>(W1, W2, w1p, w2p);

  const int B = in_sizes[0] / (12 * 128);         // 32768
  gat_fused_kernel<<<B, 128, 0, stream>>>(
      x, w1p, w2p, a_src1, a_dst1, b1, a_src2, a_dst2, b2,
      Wc1, bc1, Wc2, bc2, out);
}
